// HeteroLinear_12386685682284
// MI455X (gfx1250) — matrix operand order repaired and checked by reading, not compiled, not run on hardware
//
#include <hip/hip_runtime.h>
#include <hip/hip_bf16.h>

// HeteroLinear grouped GEMM for MI455X (gfx1250).
// out[n] = x[n] @ W[type[n]] + b[type[n]],  type_vec sorted.
// bf16 WMMA, B staged through LDS via async global->LDS copies (ASYNCcnt),
// fragments read back with ds_load_b128.

#define IN_CH 512
#define OUT_CH 512
#define KSTAGE 64                 // K elements per LDS stage
#define BCOLS 128                 // block column tile

typedef __attribute__((ext_vector_type(16))) __bf16 v16bf;
typedef __attribute__((ext_vector_type(8)))  float  v8f;

// ---------------------------------------------------------------------------
// Pre-pass: W (T, K, N) f32 row-major  ->  Wt (T, N, K) bf16  (in d_ws).
// ---------------------------------------------------------------------------
__global__ __launch_bounds__(256)
void hl_wt_transpose(const float* __restrict__ w, __bf16* __restrict__ wt, int total) {
    int idx = blockIdx.x * 256 + threadIdx.x;
    if (idx >= total) return;
    int n = idx % OUT_CH;
    int k = (idx / OUT_CH) % IN_CH;
    int t = idx / (OUT_CH * IN_CH);
    wt[(size_t)t * (IN_CH * OUT_CH) + (size_t)n * IN_CH + k] = (__bf16)w[idx];
}

// Async 16B global -> LDS copy, tracked by ASYNCcnt (no dest VGPRs).
// lds_off = group-segment byte offset (low 32 bits of the flat LDS address).
static __device__ __forceinline__ void async_g2l_b128(unsigned lds_off,
                                                      const void* gaddr) {
    asm volatile("global_load_async_to_lds_b128 %0, %1, off"
                 :: "v"(lds_off), "v"(gaddr) : "memory");
}
static __device__ __forceinline__ void wait_async0() {
    asm volatile("s_wait_asynccnt 0x0" ::: "memory");
}

// Convert 16 floats (optionally masked by s) into one A fragment (packed cvt).
template <bool MASKED>
static __device__ __forceinline__ v16bf mk_a(float4 f0, float4 f1, float4 f2,
                                             float4 f3, float s) {
    if (MASKED) {
        f0.x *= s; f0.y *= s; f0.z *= s; f0.w *= s;
        f1.x *= s; f1.y *= s; f1.z *= s; f1.w *= s;
        f2.x *= s; f2.y *= s; f2.z *= s; f2.w *= s;
        f3.x *= s; f3.y *= s; f3.z *= s; f3.w *= s;
    }
    v16bf a;
    a[0]  = (__bf16)f0.x; a[1]  = (__bf16)f0.y; a[2]  = (__bf16)f0.z; a[3]  = (__bf16)f0.w;
    a[4]  = (__bf16)f1.x; a[5]  = (__bf16)f1.y; a[6]  = (__bf16)f1.z; a[7]  = (__bf16)f1.w;
    a[8]  = (__bf16)f2.x; a[9]  = (__bf16)f2.y; a[10] = (__bf16)f2.z; a[11] = (__bf16)f2.w;
    a[12] = (__bf16)f3.x; a[13] = (__bf16)f3.y; a[14] = (__bf16)f3.z; a[15] = (__bf16)f3.w;
    return a;
}

// ---------------------------------------------------------------------------
// Main GEMM.  Block = 256 threads = 8 waves (4 in M x 2 in N).
// Block tile: 64 rows x 128 cols.  Wave tile: 16 x 64 (4x v8f accumulators).
// B tile staged in LDS: [2 stages][128 cols][64 K] bf16 = 2 x 16KB.
// ---------------------------------------------------------------------------
struct HlCtx {
    const float*  xk;             // x row for this lane (+ lane K sub-offset)
    __bf16*       sh[2];          // stage bases (shared array, keeps AS3)
    unsigned      lds[2];         // stage bases as LDS byte offsets (for async)
    int           kB;             // B lane K sub-offset (elements)
    int           colLDS;         // lane's column within the 128-col tile
    int           tid;
    int           colBlock;       // global column of tile origin
};

// Fill one stage (16KB = 1024 x 16B chunks, 4 per thread) for K range
// [kb, kb+64) of weight matrix wtt (already (N,K) bf16).
static __device__ __forceinline__ void fillB(const HlCtx& c, int s, int kb,
                                             const __bf16* __restrict__ wtt) {
    #pragma unroll
    for (int r = 0; r < 4; ++r) {
        int ch  = c.tid + 256 * r;       // 0..1023
        int col = ch >> 3;               // 0..127
        int ko  = (ch & 7) * 8;          // 0..56 elements (16B chunks)
        const __bf16* g = wtt + (size_t)(c.colBlock + col) * IN_CH + kb + ko;
        async_g2l_b128(c.lds[s] + (unsigned)(col * KSTAGE + ko) * 2u, g);
    }
}

// One 64-K stage of compute: two 32-K WMMA steps from LDS stage s.
template <bool MASKED>
static __device__ __forceinline__ void stage_compute(const HlCtx& c, int s, int kb,
                                                     float sc, v8f& acc0, v8f& acc1,
                                                     v8f& acc2, v8f& acc3) {
    const __bf16* sb = c.sh[s];
    #pragma unroll
    for (int u = 0; u < KSTAGE; u += 32) {
        float4 f0 = *(const float4*)(c.xk + kb + u + 0);
        float4 f1 = *(const float4*)(c.xk + kb + u + 4);
        float4 f2 = *(const float4*)(c.xk + kb + u + 8);
        float4 f3 = *(const float4*)(c.xk + kb + u + 12);

        const int e = c.colLDS * KSTAGE + u + c.kB;
        // LDS fragment reads through the shared array (AS3 -> ds_load_b128).
        v16bf b0 = *(const v16bf*)(sb + e + 0 * 16 * KSTAGE);
        v16bf b1 = *(const v16bf*)(sb + e + 1 * 16 * KSTAGE);
        v16bf b2 = *(const v16bf*)(sb + e + 2 * 16 * KSTAGE);
        v16bf b3 = *(const v16bf*)(sb + e + 3 * 16 * KSTAGE);

        v16bf a = mk_a<MASKED>(f0, f1, f2, f3, sc);
        acc0 = __builtin_amdgcn_wmma_f32_16x16x32_bf16(false, a, false, b0,
                                                       (short)0, acc0, false, false);
        acc1 = __builtin_amdgcn_wmma_f32_16x16x32_bf16(false, a, false, b1,
                                                       (short)0, acc1, false, false);
        acc2 = __builtin_amdgcn_wmma_f32_16x16x32_bf16(false, a, false, b2,
                                                       (short)0, acc2, false, false);
        acc3 = __builtin_amdgcn_wmma_f32_16x16x32_bf16(false, a, false, b3,
                                                       (short)0, acc3, false, false);
    }
}

// Full K loop for one weight matrix, double-buffered through LDS.
template <bool MASKED>
static __device__ __forceinline__ void kloop(const HlCtx& c, float sc,
                                             const __bf16* __restrict__ wtt,
                                             v8f& acc0, v8f& acc1,
                                             v8f& acc2, v8f& acc3) {
    constexpr int NIT = IN_CH / KSTAGE;      // 8
    fillB(c, 0, 0, wtt);
    wait_async0();
    __syncthreads();
    for (int it = 0; it < NIT; ++it) {
        const int kb = it * KSTAGE;
        if (it + 1 < NIT) fillB(c, (it + 1) & 1, kb + KSTAGE, wtt);
        stage_compute<MASKED>(c, it & 1, kb, sc, acc0, acc1, acc2, acc3);
        if (it + 1 < NIT) {
            wait_async0();       // next stage resident (this wave's copies)
            __syncthreads();     // all waves' copies visible; reads of cur done
        }
    }
    __syncthreads();             // stage buffers reusable by next kloop call
}

__global__ __launch_bounds__(256, 4)
void hl_gemm(const float* __restrict__ x,
             const int*   __restrict__ tvec,
             const float* __restrict__ bias,
             const __bf16* __restrict__ wt,
             float* __restrict__ out) {
    __shared__ __align__(64) __bf16 Bsh[2][BCOLS * KSTAGE];   // 2 x 16KB

    const int tid  = threadIdx.x;
    const int lane = tid & 31;
    const int wave = tid >> 5;
    const int wm   = wave >> 1;          // 0..3  (M sub-tile)
    const int wn   = wave & 1;           // 0..1  (N sub-tile)
    const int hi   = lane >> 4;
    const int l16  = lane & 15;

    const int blockN = blockIdx.x;       // 0..3
    const int blockM = blockIdx.y;

    const int rowBlock = blockM * 64;
    const int rowA     = rowBlock + wm * 16 + l16;
    const int colBlock = blockN * BCOLS;

    HlCtx c;
    c.tid      = tid;
    c.colBlock = colBlock;
    c.kB       = hi * 16;
    c.colLDS   = wn * 64 + l16;          // N-subtile 0 column within tile
    c.xk       = x + (size_t)rowA * IN_CH + hi * 16;
    c.sh[0]    = &Bsh[0][0];
    c.sh[1]    = &Bsh[1][0];
    c.lds[0]   = (unsigned)(size_t)&Bsh[0][0];   // flat low 32b == LDS offset
    c.lds[1]   = (unsigned)(size_t)&Bsh[1][0];

    const int tmin = tvec[rowBlock];
    const int tmax = tvec[rowBlock + 63];
    const int tA   = tvec[rowA];

    v8f acc0 = {}, acc1 = {}, acc2 = {}, acc3 = {};

    if (tmin == tmax) {
        // Fast path (4089/4096 row-tiles): single type, no masking VALU.
        kloop<false>(c, 1.0f, wt + (size_t)tmin * (IN_CH * OUT_CH),
                     acc0, acc1, acc2, acc3);
    } else {
        // Boundary tiles: one masked pass per type present (row-disjoint sums).
        for (int t = tmin; t <= tmax; ++t) {
            const float sc = (tA == t) ? 1.0f : 0.0f;
            kloop<true>(c, sc, wt + (size_t)t * (IN_CH * OUT_CH),
                        acc0, acc1, acc2, acc3);
        }
    }

    // Epilogue: C/D layout lane L, VGPR v -> row = v + 8*(L>>4), col = L&15.
    const int rowC    = rowBlock + wm * 16 + hi * 8;
    const int colBase = colBlock + wn * 64 + l16;

    int trow[8];
    #pragma unroll
    for (int v = 0; v < 8; ++v) trow[v] = tvec[rowC + v];

    #pragma unroll
    for (int v = 0; v < 8; ++v) {
        float* __restrict__ orow = out + (size_t)(rowC + v) * OUT_CH;
        const float* __restrict__ brow = bias + (size_t)trow[v] * OUT_CH;
        orow[colBase +  0] = acc0[v] + brow[colBase +  0];
        orow[colBase + 16] = acc1[v] + brow[colBase + 16];
        orow[colBase + 32] = acc2[v] + brow[colBase + 32];
        orow[colBase + 48] = acc3[v] + brow[colBase + 48];
    }
}

// ---------------------------------------------------------------------------
extern "C" void kernel_launch(void* const* d_in, const int* in_sizes, int n_in,
                              void* d_out, int out_size, void* d_ws, size_t ws_size,
                              hipStream_t stream) {
    const float* x    = (const float*)d_in[0];
    const int*   tvec = (const int*)  d_in[1];
    const float* w    = (const float*)d_in[2];
    const float* bias = (const float*)d_in[3];
    float*       out  = (float*)d_out;
    __bf16*      wt   = (__bf16*)d_ws;   // needs T*IN_CH*OUT_CH*2 = 4MB scratch

    const int Nrows = in_sizes[0] / IN_CH;
    const int T     = in_sizes[2] / (IN_CH * OUT_CH);
    const int total = T * IN_CH * OUT_CH;

    hl_wt_transpose<<<(total + 255) / 256, 256, 0, stream>>>(w, wt, total);

    dim3 grid(OUT_CH / BCOLS, Nrows / 64);
    hl_gemm<<<grid, 256, 0, stream>>>(x, tvec, bias, wt, out);
}
